// DictionaryLearning_83313775608037
// MI455X (gfx1250) — matrix operand order repaired and checked by reading, not compiled, not run on hardware
//
#include <hip/hip_runtime.h>
#include <math.h>

typedef _Float16 f16;
typedef __attribute__((ext_vector_type(16))) _Float16 v16h;
typedef __attribute__((ext_vector_type(8)))  _Float16 v8h;
typedef __attribute__((ext_vector_type(4)))  _Float16 v4h;
typedef __attribute__((ext_vector_type(8)))  float    v8f;
typedef __attribute__((ext_vector_type(4)))  float    v4f;

#define ATOM_DIM   1024   // M = C*P*P = 64*16
#define N_ATOMS    512
#define N_PATCHES  32768  // 32 * 32 * 32
#define K_MAX      5

// ---------------------------------------------------------------------------
// 1) Normalize dictionary columns; emit Dn^T in f32 (recon) and f16 (WMMA).
//    Din: [1024, 512] row-major.  DnT: [512, 1024] row-major.
// ---------------------------------------------------------------------------
__global__ __launch_bounds__(256) void normdict_kernel(
    const float* __restrict__ Din, float* __restrict__ DnT32, f16* __restrict__ DnT16)
{
    __shared__ float red[256];
    const int n = blockIdx.x;     // column / atom index
    const int t = threadIdx.x;
    float vals[4];
    float s = 0.f;
#pragma unroll
    for (int i = 0; i < 4; ++i) {
        float v = Din[(size_t)(t + i * 256) * N_ATOMS + n];
        vals[i] = v;
        s += v * v;
    }
    red[t] = s;
    __syncthreads();
    for (int w = 128; w > 0; w >>= 1) {
        if (t < w) red[t] += red[t + w];
        __syncthreads();
    }
    const float scale = 1.0f / (sqrtf(red[0]) + 1e-10f);
#pragma unroll
    for (int i = 0; i < 4; ++i) {
        const int m = t + i * 256;
        const float v = vals[i] * scale;
        DnT32[(size_t)n * ATOM_DIM + m] = v;
        DnT16[(size_t)n * ATOM_DIM + m] = (f16)v;
    }
}

// ---------------------------------------------------------------------------
// 2) Extract 4x4 patches: z [32,64,128,128] -> patches f16 [32768, 1024].
//    Feature index m = c*16 + ph*4 + pw  (pw contiguous -> float4 reads).
// ---------------------------------------------------------------------------
__global__ __launch_bounds__(256) void extract_kernel(
    const float* __restrict__ z, f16* __restrict__ patches)
{
    const int p = blockIdx.x;
    const int t = threadIdx.x;
    const int b  = p >> 10;
    const int hb = (p >> 5) & 31;
    const int wb = p & 31;
    const int c  = t >> 2;
    const int ph = t & 3;
    const size_t zoff = (((size_t)(b * 64 + c) * 128) + (size_t)(hb * 4 + ph)) * 128 + (size_t)(wb * 4);
    const v4f v = *(const v4f*)(z + zoff);
    v4h o;
#pragma unroll
    for (int i = 0; i < 4; ++i) o[i] = (f16)v[i];
    *(v4h*)(patches + (size_t)p * ATOM_DIM + t * 4) = o;
}

// ---------------------------------------------------------------------------
// 3) f16 WMMA GEMM: C[M,N] = A[M,K] * B[K,N], B given transposed (BT [N,K]).
//    One wave computes a 32(M) x 64(N) strip: 2 A-fragments x 4 B-fragments,
//    8 WMMAs per 12 b128 loads (B reused across both A fragments).
//    Fragment layouts used here (wave32, contiguous k order, same for A and B):
//      A 16x32 f16: lanes0-15 row=lane, K k0..k0+15 (contiguous);
//                   lanes16-31 row=lane-16, K k0+16..k0+31.
//      B 32x16 f16: lanes0-15 col=lane, K k0..k0+15 (contiguous);
//                   lanes16-31 col=lane-16, K k0+16..k0+31.
//      C/D 16x16 f32: VGPR v -> row v (lanes0-15) / 8+v (lanes16-31), col=lane&15.
// ---------------------------------------------------------------------------
__global__ __launch_bounds__(256) void gemm_f16_wmma(
    const f16* __restrict__ A, const f16* __restrict__ BT, float* __restrict__ C,
    int Mrows, int Ncols, int K, float diag_add)
{
    const int wave = blockIdx.x * (blockDim.x >> 5) + (threadIdx.x >> 5);
    const int lane = threadIdx.x & 31;
    const int ngrp = Ncols >> 6;                 // groups of 4 N-tiles
    const int tm = wave / ngrp;                  // 32-row strip index
    const int tg = wave % ngrp;                  // 64-col group index
    if (tm * 32 >= Mrows) return;
    const int  r  = lane & 15;
    const bool hi = lane >= 16;

    const f16* aBase0 = A  + (size_t)(tm * 32 + r) * K + (hi ? 16 : 0);
    const f16* aBase1 = aBase0 + (size_t)16 * K;
    const f16* bCol   = BT + (size_t)(tg * 64 + r) * K + (hi ? 16 : 0);

    v8f acc[8] = {};    // acc[j] = (m-tile 0, n-tile j), acc[4+j] = (m-tile 1, n-tile j)

    for (int k0 = 0; k0 < K; k0 += 32) {
        const v8h a00 = *(const v8h*)(aBase0 + k0);
        const v8h a01 = *(const v8h*)(aBase0 + k0 + 8);
        const v8h a10 = *(const v8h*)(aBase1 + k0);
        const v8h a11 = *(const v8h*)(aBase1 + k0 + 8);
        const v16h af0 = __builtin_shufflevector(a00, a01,
            0, 1, 2, 3, 4, 5, 6, 7, 8, 9, 10, 11, 12, 13, 14, 15);
        const v16h af1 = __builtin_shufflevector(a10, a11,
            0, 1, 2, 3, 4, 5, 6, 7, 8, 9, 10, 11, 12, 13, 14, 15);
        __builtin_prefetch(aBase0 + k0 + 64, 0, 1);   // global_prefetch_b8
        __builtin_prefetch(aBase1 + k0 + 64, 0, 1);

#pragma unroll
        for (int j = 0; j < 4; ++j) {
            const v16h bf = *(const v16h*)(bCol + (size_t)(16 * j) * K + k0);
            acc[j]     = __builtin_amdgcn_wmma_f32_16x16x32_f16(false, af0, false, bf, (short)0, acc[j],     false, false);
            acc[4 + j] = __builtin_amdgcn_wmma_f32_16x16x32_f16(false, af1, false, bf, (short)0, acc[4 + j], false, false);
        }
    }

#pragma unroll
    for (int mi = 0; mi < 2; ++mi) {
        const int rowBase = tm * 32 + mi * 16 + (hi ? 8 : 0);
#pragma unroll
        for (int j = 0; j < 4; ++j) {
            const v8f a = acc[mi * 4 + j];
            const int gcol = tg * 64 + j * 16 + r;
#pragma unroll
            for (int v = 0; v < 8; ++v) {
                const int grow = rowBase + v;
                float val = a[v];
                if (diag_add != 0.0f && grow == gcol) val += diag_add;
                C[(size_t)grow * Ncols + gcol] = val;
            }
        }
    }
}

// ---------------------------------------------------------------------------
// 4) Batch-OMP: one wave32 per patch. Lane owns atoms n = t*32+lane (t<16).
//    5 greedy steps: masked argmax -> progressive Cholesky -> tri solves ->
//    h = h_bar - x . G[I].  Writes 5 (coeff, index) pairs per patch.
// ---------------------------------------------------------------------------
__global__ __launch_bounds__(256) void omp_kernel(
    const float* __restrict__ h_bar, const float* __restrict__ G,
    float* __restrict__ cx, int* __restrict__ cI, int P)
{
    const int wave = blockIdx.x * (blockDim.x >> 5) + (threadIdx.x >> 5);
    const int lane = threadIdx.x & 31;
    if (wave >= P) return;

    const float* hb = h_bar + (size_t)wave * N_ATOMS;
    float hbar_loc[16], h_loc[16];
#pragma unroll
    for (int t = 0; t < 16; ++t) { hbar_loc[t] = hb[t * 32 + lane]; h_loc[t] = hbar_loc[t]; }

    unsigned sel = 0;          // bit t <=> atom t*32+lane already selected
    int   I[K_MAX];
    float xs[K_MAX];
    float hs[K_MAX];
    float Lm[15];              // lower-tri, row-major: Lm[r*(r+1)/2 + c]

    for (int k = 0; k < K_MAX; ++k) {
        // masked argmax of |h| (tie -> lowest index, like jnp.argmax)
        float best = -1.0f; int bestn = N_ATOMS;
#pragma unroll
        for (int t = 0; t < 16; ++t) {
            const float s = ((sel >> t) & 1u) ? -1.0f : fabsf(h_loc[t]);
            const int n = t * 32 + lane;
            if (s > best || (s == best && n < bestn)) { best = s; bestn = n; }
        }
#pragma unroll
        for (int off = 16; off >= 1; off >>= 1) {
            const float ov = __shfl_xor(best, off, 32);
            const int   on = __shfl_xor(bestn, off, 32);
            if (ov > best || (ov == best && on < bestn)) { best = ov; bestn = on; }
        }
        const int idx = bestn;
        if ((idx & 31) == lane) sel |= 1u << (idx >> 5);
        I[k] = idx;

        // progressive Cholesky row (all lanes redundantly; G reads broadcast)
        if (k > 0) {
            float w[K_MAX - 1];
            for (int j = 0; j < k; ++j) {
                float g = G[(size_t)I[j] * N_ATOMS + idx];
                for (int c = 0; c < j; ++c) g -= Lm[j * (j + 1) / 2 + c] * w[c];
                w[j] = g / Lm[j * (j + 1) / 2 + j];
            }
            float s2 = 0.f;
            for (int j = 0; j < k; ++j) s2 += w[j] * w[j];
            const float corner = sqrtf(fmaxf(1.0f - s2, 1e-5f));
            for (int j = 0; j < k; ++j) Lm[k * (k + 1) / 2 + j] = w[j];
            Lm[k * (k + 1) / 2 + k] = corner;
        } else {
            Lm[0] = 1.0f;
        }

        hs[k] = hb[idx];
        // forward solve L y = hs
        float y[K_MAX];
        for (int rr = 0; rr <= k; ++rr) {
            float v = hs[rr];
            for (int c = 0; c < rr; ++c) v -= Lm[rr * (rr + 1) / 2 + c] * y[c];
            y[rr] = v / Lm[rr * (rr + 1) / 2 + rr];
        }
        // back solve L^T x = y
        for (int rr = k; rr >= 0; --rr) {
            float v = y[rr];
            for (int j = rr + 1; j <= k; ++j) v -= Lm[j * (j + 1) / 2 + rr] * xs[j];
            xs[rr] = v / Lm[rr * (rr + 1) / 2 + rr];
        }

        // h = h_bar - x . G[I]  (coalesced G row reads across lanes)
        if (k < K_MAX - 1) {
#pragma unroll
            for (int t = 0; t < 16; ++t) {
                const int n = t * 32 + lane;
                float acc = 0.f;
                for (int j = 0; j <= k; ++j) acc += xs[j] * G[(size_t)I[j] * N_ATOMS + n];
                h_loc[t] = hbar_loc[t] - acc;
            }
        }
    }

    if (lane < K_MAX) {
        cx[(size_t)wave * K_MAX + lane] = xs[lane];
        cI[(size_t)wave * K_MAX + lane] = I[lane];
    }
}

// ---------------------------------------------------------------------------
// 5) Sparse reconstruction: out[., p] = sum_k x_k * Dn[:, I_k], scattered NCHW.
// ---------------------------------------------------------------------------
__global__ __launch_bounds__(256) void recon_kernel(
    const float* __restrict__ DnT32, const float* __restrict__ cx,
    const int* __restrict__ cI, float* __restrict__ out)
{
    __shared__ float sx[K_MAX];
    __shared__ int   sI[K_MAX];
    const int p = blockIdx.x;
    const int t = threadIdx.x;
    if (t < K_MAX) { sx[t] = cx[(size_t)p * K_MAX + t]; sI[t] = cI[(size_t)p * K_MAX + t]; }
    __syncthreads();

    const int m = t * 4;
    v4f acc = {0.f, 0.f, 0.f, 0.f};
#pragma unroll
    for (int j = 0; j < K_MAX; ++j) {
        const v4f d = *(const v4f*)(DnT32 + (size_t)sI[j] * ATOM_DIM + m);
        const float xv = sx[j];
#pragma unroll
        for (int i = 0; i < 4; ++i) acc[i] += xv * d[i];
    }
    const int b  = p >> 10;
    const int hb = (p >> 5) & 31;
    const int wb = p & 31;
    const int c  = t >> 2;
    const int ph = t & 3;
    const size_t zoff = (((size_t)(b * 64 + c) * 128) + (size_t)(hb * 4 + ph)) * 128 + (size_t)(wb * 4);
    *(v4f*)(out + zoff) = acc;
}

// ---------------------------------------------------------------------------
extern "C" void kernel_launch(void* const* d_in, const int* in_sizes, int n_in,
                              void* d_out, int out_size, void* d_ws, size_t ws_size,
                              hipStream_t stream)
{
    const float* z    = (const float*)d_in[0];   // [32,64,128,128]
    const float* dict = (const float*)d_in[1];   // [1024,512]
    float* out = (float*)d_out;

    char* ws = (char*)d_ws;
    size_t off = 0;
    auto carve = [&](size_t bytes) -> void* {
        void* p = ws + off;
        off += (bytes + 255) & ~(size_t)255;
        return p;
    };
    f16*   patches = (f16*)  carve((size_t)N_PATCHES * ATOM_DIM * sizeof(f16));   // 67 MB
    float* h_bar   = (float*)carve((size_t)N_PATCHES * N_ATOMS * sizeof(float));  // 67 MB
    f16*   DnT16   = (f16*)  carve((size_t)N_ATOMS * ATOM_DIM * sizeof(f16));     //  1 MB
    float* DnT32   = (float*)carve((size_t)N_ATOMS * ATOM_DIM * sizeof(float));   //  2 MB
    float* G       = (float*)carve((size_t)N_ATOMS * N_ATOMS * sizeof(float));    //  1 MB
    float* cx      = (float*)carve((size_t)N_PATCHES * K_MAX * sizeof(float));
    int*   cI      = (int*)  carve((size_t)N_PATCHES * K_MAX * sizeof(int));
    (void)ws_size; (void)in_sizes; (void)n_in; (void)out_size;

    // 1) normalize dictionary -> DnT (f32 + f16)
    normdict_kernel<<<N_ATOMS, 256, 0, stream>>>(dict, DnT32, DnT16);
    // 2) patch extraction -> f16
    extract_kernel<<<N_PATCHES, 256, 0, stream>>>(z, patches);
    // 3a) G = Dn^T Dn + eps*I (512x512, K=1024):
    //     waves = (512/32) * (512/64) = 16 * 8 = 128 -> 16 blocks of 8 waves
    gemm_f16_wmma<<<16, 256, 0, stream>>>(DnT16, DnT16, G, N_ATOMS, N_ATOMS, ATOM_DIM, 1e-5f);
    // 3b) h_bar = patches @ Dn (32768x512, K=1024):
    //     waves = (32768/32) * 8 = 8192 -> 1024 blocks of 8 waves
    gemm_f16_wmma<<<1024, 256, 0, stream>>>(patches, DnT16, h_bar, N_PATCHES, N_ATOMS, ATOM_DIM, 0.0f);
    // 4) batch OMP (one wave per patch)
    omp_kernel<<<N_PATCHES / 8, 256, 0, stream>>>(h_bar, G, cx, cI, N_PATCHES);
    // 5) sparse reconstruction + patch reassembly
    recon_kernel<<<N_PATCHES, 256, 0, stream>>>(DnT32, cx, cI, out);
}
